// DiverseAsmaaMoE_26362509263285
// MI455X (gfx1250) — compile-verified, matrix operand order checked by reading, not run on hardware
//
#include <hip/hip_runtime.h>
#include <hip/hip_bf16.h>

// ---------------------------------------------------------------------------
// MoE top-2 forward for MI455X (gfx1250, wave32, WMMA bf16 16x16x32).
// B=2 S=1024 D=1024 E=14 F=2048 ; N = B*S = 2048 tokens.
//
// Roofline: top-2-only work ~34 GFLOP; fp32 weights 235 MB -> ~10us @23.3TB/s;
// bf16 WMMA compute ~10-15us. Weights converted fp32->bf16 inline in LDS
// (no separate pass); per-expert weight slab (8.4 MB) stays L2-resident
// across M-tiles. Double-buffered LDS, one barrier per K-step, 2 WMMAs per
// wave per K-step, all staging loads unconditional b128, B-transpose staged
// as cvt_pk pairs -> ds_store_b32 (no 16-bit extract churn).
// ---------------------------------------------------------------------------

typedef __bf16 bf16_t;
typedef __bf16 v16bf __attribute__((ext_vector_type(16)));
typedef __bf16 v8bf  __attribute__((ext_vector_type(8)));
typedef __bf16 v4bf  __attribute__((ext_vector_type(4)));
typedef __bf16 v2bf  __attribute__((ext_vector_type(2)));
typedef float  v8f   __attribute__((ext_vector_type(8)));
typedef float  v4f   __attribute__((ext_vector_type(4)));

#define TOKENS 2048
#define DMODEL 1024
#define FFN    2048
#define NEXP   14

#define BM 32          // rows (tokens) per block tile
#define BN 128         // cols per block tile
#define BK 32          // K per step
#define APITCH 40      // LDS row pitch (bf16 elems): 80B -> 16B-aligned chunks
#define BPITCH 40

// Workspace layout (bytes, all 256B aligned)
#define CNT_OFF   0ull                                        // 14 ints
#define LIST_OFF  1024ull                                     // E*TOKENS ints
#define WSLOT_OFF (LIST_OFF + (size_t)NEXP * TOKENS * 4)      // 2*TOKENS floats
#define HH_OFF    (WSLOT_OFF + (size_t)2 * TOKENS * 4)        // 2*TOKENS*FFN bf16
#define YOUT_OFF  (HH_OFF + (size_t)2 * TOKENS * FFN * 2)     // 2*TOKENS*DMODEL f32
// total ~= 33.7 MB

// ---------------------------------------------------------------------------
__global__ __launch_bounds__(32) void moe1250_init(int* __restrict__ cnt) {
    if (threadIdx.x < NEXP) cnt[threadIdx.x] = 0;
}

// ---------------------------------------------------------------------------
// Router: one wave32 per token. logits = x . gate_w^T, softmax, top-2.
__global__ __launch_bounds__(256) void moe1250_router(
    const float* __restrict__ x, const float* __restrict__ gate_w,
    int* __restrict__ cnt, int* __restrict__ list, float* __restrict__ wslot)
{
    const int wave = threadIdx.x >> 5;
    const int lane = threadIdx.x & 31;
    const int tok  = blockIdx.x * 8 + wave;
    const float* xr = x + (size_t)tok * DMODEL;

    float acc[NEXP];
#pragma unroll
    for (int e = 0; e < NEXP; ++e) acc[e] = 0.f;

    for (int i = 0; i < DMODEL / 32; ++i) {
        const int d = i * 32 + lane;
        const float xv = xr[d];
#pragma unroll
        for (int e = 0; e < NEXP; ++e)
            acc[e] += xv * gate_w[e * DMODEL + d];
    }
#pragma unroll
    for (int e = 0; e < NEXP; ++e) {
#pragma unroll
        for (int off = 16; off > 0; off >>= 1)
            acc[e] += __shfl_xor(acc[e], off, 32);
    }

    if (lane == 0) {
        float mx = acc[0];
#pragma unroll
        for (int e = 1; e < NEXP; ++e) mx = fmaxf(mx, acc[e]);
        float p[NEXP]; float se = 0.f;
#pragma unroll
        for (int e = 0; e < NEXP; ++e) { p[e] = __expf(acc[e] - mx); se += p[e]; }
        const float inv = 1.f / se;
#pragma unroll
        for (int e = 0; e < NEXP; ++e) p[e] *= inv;

        int i0 = 0;
#pragma unroll
        for (int e = 1; e < NEXP; ++e) if (p[e] > p[i0]) i0 = e;
        int i1 = (i0 == 0) ? 1 : 0;
#pragma unroll
        for (int e = 0; e < NEXP; ++e) if (e != i0 && p[e] > p[i1]) i1 = e;

        const float s  = fmaxf(p[i0] + p[i1], 1e-10f);
        const float w0 = p[i0] / s, w1 = p[i1] / s;

        const int pos0 = atomicAdd(&cnt[i0], 1);
        list[i0 * TOKENS + pos0] = 2 * tok;
        wslot[2 * tok] = w0;
        const int pos1 = atomicAdd(&cnt[i1], 1);
        list[i1 * TOKENS + pos1] = 2 * tok + 1;
        wslot[2 * tok + 1] = w1;
    }
}

// ---------------------------------------------------------------------------
// Fragment gather matching the ISA 16-bit A/B 16x32 VGPR layout:
// lane holds row (m or n) = lane&15, K elements {kb..kb+7} and {16+kb..16+kb+7}
// with kb = (lane>>4)*8. Row pitch 40 bf16 = 80B keeps both chunks 16B-aligned
// -> two ds_load_b128 per fragment.
__device__ __forceinline__ v16bf moe1250_frag(const bf16_t* __restrict__ row, int lane)
{
    const int kb = (lane >> 4) << 3;
    const v8bf lo = *(const v8bf*)(row + kb);
    const v8bf hi = *(const v8bf*)(row + 16 + kb);
    v16bf r;
#pragma unroll
    for (int i = 0; i < 8; ++i) { r[i] = lo[i]; r[i + 8] = hi[i]; }
    return r;
}

__device__ __forceinline__ v4bf moe1250_cvt4(v4f v)
{
    v4bf r;
#pragma unroll
    for (int i = 0; i < 4; ++i) r[i] = (bf16_t)v[i];
    return r;
}

// Store the B tile transposed: thread holds two ADJACENT k rows (k, k+1) for
// 4 consecutive n. Each (n, k-pair) becomes one v_cvt_pk_bf16_f32 + one
// 4-byte-aligned ds_store_b32 -- no 16-bit extraction moves.
__device__ __forceinline__ void moe1250_storeB(
    bf16_t* __restrict__ bbase /* &b_lds[buf][0][0] */,
    int nn4, int kk, v4f r0 /* row k */, v4f r1 /* row k+1 */)
{
#pragma unroll
    for (int q = 0; q < 4; ++q) {
        v2bf p;
        p[0] = (bf16_t)r0[q];
        p[1] = (bf16_t)r1[q];
        *(v2bf*)(bbase + (size_t)(nn4 + q) * BPITCH + kk) = p;
    }
}

// ---------------------------------------------------------------------------
// Block: 256 thr = 8 waves as 2(M) x 4(N); each wave owns one 16x32 strip ->
// 2 WMMAs per K-step. Double-buffered LDS, one barrier per K-step.

// GEMM1: hh[slot, 0:F] = gelu( x[tok] @ w1[e] + b1[e] )   (bf16 out)
__global__ __launch_bounds__(256) void moe1250_gemm1(
    const float* __restrict__ x, const float* __restrict__ w1,
    const float* __restrict__ b1, const int* __restrict__ cnt,
    const int* __restrict__ list, bf16_t* __restrict__ hh)
{
    __shared__ bf16_t a_lds[2][BM][APITCH];
    __shared__ bf16_t b_lds[2][BN][BPITCH];
    __shared__ int    slist[BM];

    const int e  = blockIdx.x;
    const int n0 = blockIdx.y * BN;
    const int Ne = cnt[e];
    const int mtiles = (Ne + BM - 1) / BM;
    const float* __restrict__ wE = w1 + (size_t)e * DMODEL * FFN;

    const int tid  = threadIdx.x;
    const int lane = tid & 31;
    const int wave = tid >> 5;
    const int wm   = wave >> 2;     // 0..1
    const int wn   = wave & 3;      // 0..3

    // A staging: row = tid>>3 (0..31), 4 consecutive k
    const int arow = tid >> 3;
    const int acol = (tid & 7) * 4;
    // B staging: k-pairs (kk, kk+1) with kk = (tid>>5)*2 + 16*j, 4 consecutive n
    const int bk2  = (tid >> 5) * 2;    // 0,2,..,14
    const int bn4  = (tid & 31) * 4;    // n offset 0..124

    const int nglob0 = n0 + wn * 32 + (lane & 15);
    const float b1v0 = b1[e * FFN + nglob0];
    const float b1v1 = b1[e * FFN + nglob0 + 16];

    for (int mt = 0; mt < mtiles; ++mt) {
        const int rows = min(BM, Ne - mt * BM);
        // clamped publish: padded rows alias row rows-1; their outputs are
        // computed (harmlessly) and never stored -> unconditional A loads.
        if (tid < BM) slist[tid] = list[e * TOKENS + mt * BM + min(tid, rows - 1)];
        __syncthreads();

        const float* aptr = x + (size_t)(slist[arow] >> 1) * DMODEL + acol;

        // ---- prologue: stage k0 = 0 into buffer 0
        {
            const v4f av = *(const v4f*)aptr;
            *(v4bf*)&a_lds[0][arow][acol] = moe1250_cvt4(av);
#pragma unroll
            for (int j = 0; j < 2; ++j) {
                const int kk = bk2 + 16 * j;
                const v4f r0 = *(const v4f*)&wE[(size_t)kk * FFN + n0 + bn4];
                const v4f r1 = *(const v4f*)&wE[(size_t)(kk + 1) * FFN + n0 + bn4];
                moe1250_storeB(&b_lds[0][0][0], bn4, kk, r0, r1);
            }
        }
        __syncthreads();

        v8f c0 = {}, c1 = {};
        int cur = 0;
        for (int ki = 0; ki < DMODEL / BK; ++ki) {
            const int k0 = ki * BK;
            const bool have_next = (k0 + BK) < DMODEL;

            // issue next tile's global loads (in flight during WMMA)
            v4f av, br0[2], br1[2];
            if (have_next) {
                av = *(const v4f*)(aptr + k0 + BK);
#pragma unroll
                for (int j = 0; j < 2; ++j) {
                    const int kk = k0 + BK + bk2 + 16 * j;
                    br0[j] = *(const v4f*)&wE[(size_t)kk * FFN + n0 + bn4];
                    br1[j] = *(const v4f*)&wE[(size_t)(kk + 1) * FFN + n0 + bn4];
                }
                if (tid < 32 && k0 + 2 * BK < DMODEL)
                    __builtin_prefetch(&wE[(size_t)(k0 + 2 * BK + lane) * FFN + n0], 0, 1);
            }

            // compute from current buffer
            {
                const v16bf a  = moe1250_frag(&a_lds[cur][wm * 16 + (lane & 15)][0], lane);
                const v16bf bA = moe1250_frag(&b_lds[cur][wn * 32 + (lane & 15)][0], lane);
                const v16bf bB = moe1250_frag(&b_lds[cur][wn * 32 + 16 + (lane & 15)][0], lane);
                c0 = __builtin_amdgcn_wmma_f32_16x16x32_bf16(false, a, false, bA,
                                                             (short)0, c0, false, false);
                c1 = __builtin_amdgcn_wmma_f32_16x16x32_bf16(false, a, false, bB,
                                                             (short)0, c1, false, false);
            }

            // convert + store next tile into the other buffer
            if (have_next) {
                const int nxt = cur ^ 1;
                *(v4bf*)&a_lds[nxt][arow][acol] = moe1250_cvt4(av);
#pragma unroll
                for (int j = 0; j < 2; ++j)
                    moe1250_storeB(&b_lds[nxt][0][0], bn4, bk2 + 16 * j, br0[j], br1[j]);
            }
            __syncthreads();
            cur ^= 1;
        }

        // epilogue: C layout: vgpr r, lanes 0-15 -> M=r ; lanes 16-31 -> M=8+r
#pragma unroll
        for (int r = 0; r < 8; ++r) {
            const int mloc = wm * 16 + r + ((lane >> 4) << 3);
            if (mloc < rows) {
                const int slot = slist[mloc];
                const float v0 = c0[r] + b1v0;
                const float v1 = c1[r] + b1v1;
                const float g0 = 0.5f * v0 * (1.0f + erff(v0 * 0.70710678118654752f));
                const float g1 = 0.5f * v1 * (1.0f + erff(v1 * 0.70710678118654752f));
                hh[(size_t)slot * FFN + nglob0]      = (bf16_t)g0;
                hh[(size_t)slot * FFN + nglob0 + 16] = (bf16_t)g1;
            }
        }
        __syncthreads();
    }
}

// GEMM2: yout[slot, 0:D] = hh[slot] @ w2[e] + b2[e]   (fp32 out)
__global__ __launch_bounds__(256) void moe1250_gemm2(
    const bf16_t* __restrict__ hh, const float* __restrict__ w2,
    const float* __restrict__ b2, const int* __restrict__ cnt,
    const int* __restrict__ list, float* __restrict__ yout)
{
    __shared__ bf16_t a_lds[2][BM][APITCH];
    __shared__ bf16_t b_lds[2][BN][BPITCH];
    __shared__ int    slist[BM];

    const int e  = blockIdx.x;
    const int n0 = blockIdx.y * BN;
    const int Ne = cnt[e];
    const int mtiles = (Ne + BM - 1) / BM;
    const float* __restrict__ wE = w2 + (size_t)e * FFN * DMODEL;

    const int tid  = threadIdx.x;
    const int lane = tid & 31;
    const int wave = tid >> 5;
    const int wm   = wave >> 2;
    const int wn   = wave & 3;

    const int arow = tid >> 3;
    const int acol = (tid & 7) * 4;
    const int bk2  = (tid >> 5) * 2;
    const int bn4  = (tid & 31) * 4;

    const int nglob0 = n0 + wn * 32 + (lane & 15);
    const float b2v0 = b2[e * DMODEL + nglob0];
    const float b2v1 = b2[e * DMODEL + nglob0 + 16];

    for (int mt = 0; mt < mtiles; ++mt) {
        const int rows = min(BM, Ne - mt * BM);
        if (tid < BM) slist[tid] = list[e * TOKENS + mt * BM + min(tid, rows - 1)];
        __syncthreads();

        const bf16_t* aptr = hh + (size_t)slist[arow] * FFN + acol;

        {
            *(v4bf*)&a_lds[0][arow][acol] = *(const v4bf*)aptr;
#pragma unroll
            for (int j = 0; j < 2; ++j) {
                const int kk = bk2 + 16 * j;
                const v4f r0 = *(const v4f*)&wE[(size_t)kk * DMODEL + n0 + bn4];
                const v4f r1 = *(const v4f*)&wE[(size_t)(kk + 1) * DMODEL + n0 + bn4];
                moe1250_storeB(&b_lds[0][0][0], bn4, kk, r0, r1);
            }
        }
        __syncthreads();

        v8f c0 = {}, c1 = {};
        int cur = 0;
        for (int ki = 0; ki < FFN / BK; ++ki) {
            const int k0 = ki * BK;
            const bool have_next = (k0 + BK) < FFN;

            v4bf av; v4f br0[2], br1[2];
            if (have_next) {
                av = *(const v4bf*)(aptr + k0 + BK);
#pragma unroll
                for (int j = 0; j < 2; ++j) {
                    const int kk = k0 + BK + bk2 + 16 * j;
                    br0[j] = *(const v4f*)&wE[(size_t)kk * DMODEL + n0 + bn4];
                    br1[j] = *(const v4f*)&wE[(size_t)(kk + 1) * DMODEL + n0 + bn4];
                }
                if (tid < 32 && k0 + 2 * BK < FFN)
                    __builtin_prefetch(&wE[(size_t)(k0 + 2 * BK + lane) * DMODEL + n0], 0, 1);
            }

            {
                const v16bf a  = moe1250_frag(&a_lds[cur][wm * 16 + (lane & 15)][0], lane);
                const v16bf bA = moe1250_frag(&b_lds[cur][wn * 32 + (lane & 15)][0], lane);
                const v16bf bB = moe1250_frag(&b_lds[cur][wn * 32 + 16 + (lane & 15)][0], lane);
                c0 = __builtin_amdgcn_wmma_f32_16x16x32_bf16(false, a, false, bA,
                                                             (short)0, c0, false, false);
                c1 = __builtin_amdgcn_wmma_f32_16x16x32_bf16(false, a, false, bB,
                                                             (short)0, c1, false, false);
            }

            if (have_next) {
                const int nxt = cur ^ 1;
                *(v4bf*)&a_lds[nxt][arow][acol] = av;
#pragma unroll
                for (int j = 0; j < 2; ++j)
                    moe1250_storeB(&b_lds[nxt][0][0], bn4, bk2 + 16 * j, br0[j], br1[j]);
            }
            __syncthreads();
            cur ^= 1;
        }

#pragma unroll
        for (int r = 0; r < 8; ++r) {
            const int mloc = wm * 16 + r + ((lane >> 4) << 3);
            if (mloc < rows) {
                const int slot = slist[mloc];
                yout[(size_t)slot * DMODEL + nglob0]      = c0[r] + b2v0;
                yout[(size_t)slot * DMODEL + nglob0 + 16] = c1[r] + b2v1;
            }
        }
        __syncthreads();
    }
}

// ---------------------------------------------------------------------------
// Combine: out = clip(x + w0*y[2t] + w1*y[2t+1], -100, 100), float4 per thread
__global__ __launch_bounds__(256) void moe1250_combine(
    const float* __restrict__ x, const float* __restrict__ wslot,
    const float* __restrict__ yout, float* __restrict__ out)
{
    const int gid  = blockIdx.x * 256 + threadIdx.x;
    const int base = gid * 4;
    const int tok  = base >> 10;           // / DMODEL
    const int d    = base & (DMODEL - 1);
    const float w0 = wslot[2 * tok];
    const float w1 = wslot[2 * tok + 1];
    const v4f xv = *(const v4f*)(x + base);
    const v4f y0 = *(const v4f*)(yout + (size_t)(2 * tok) * DMODEL + d);
    const v4f y1 = *(const v4f*)(yout + (size_t)(2 * tok + 1) * DMODEL + d);
    v4f o;
#pragma unroll
    for (int i = 0; i < 4; ++i) {
        const float v = xv[i] + w0 * y0[i] + w1 * y1[i];
        o[i] = fminf(fmaxf(v, -100.f), 100.f);
    }
    *(v4f*)(out + base) = o;
}

// ---------------------------------------------------------------------------
extern "C" void kernel_launch(void* const* d_in, const int* in_sizes, int n_in,
                              void* d_out, int out_size, void* d_ws, size_t ws_size,
                              hipStream_t stream)
{
    (void)in_sizes; (void)n_in; (void)out_size; (void)ws_size;

    const float* h      = (const float*)d_in[0];  // [2,1024,1024]
    const float* gate_w = (const float*)d_in[1];  // [14,1024]
    const float* w1     = (const float*)d_in[2];  // [14,1024,2048]
    const float* b1     = (const float*)d_in[3];  // [14,2048]
    const float* w2     = (const float*)d_in[4];  // [14,2048,1024]
    const float* b2     = (const float*)d_in[5];  // [14,1024]
    float* out = (float*)d_out;

    char* ws = (char*)d_ws;
    int*    cnt   = (int*)(ws + CNT_OFF);
    int*    list  = (int*)(ws + LIST_OFF);
    float*  wslot = (float*)(ws + WSLOT_OFF);
    bf16_t* hh    = (bf16_t*)(ws + HH_OFF);
    float*  yout  = (float*)(ws + YOUT_OFF);

    moe1250_init<<<1, 32, 0, stream>>>(cnt);
    moe1250_router<<<TOKENS / 8, 256, 0, stream>>>(h, gate_w, cnt, list, wslot);
    moe1250_gemm1<<<dim3(NEXP, FFN / BN), 256, 0, stream>>>(h, w1, b1, cnt, list, hh);
    moe1250_gemm2<<<dim3(NEXP, DMODEL / BN), 256, 0, stream>>>(hh, w2, b2, cnt, list, yout);
    moe1250_combine<<<(TOKENS * DMODEL) / 1024, 256, 0, stream>>>(h, wslot, yout, out);
}
